// Attention_19877108646354
// MI455X (gfx1250) — matrix operand order repaired and checked by reading, not compiled, not run on hardware
//
#include <hip/hip_runtime.h>
#include <hip/hip_bf16.h>
#include <math.h>
#include <stdint.h>

// Problem dims (fixed by the reference)
#define BB 32
#define SS 2048
#define HH 1024
#define AA 256

// WMMA vector types (gfx1250 CDNA5)
typedef __attribute__((ext_vector_type(16))) __bf16 v16bf;
typedef __attribute__((ext_vector_type(8)))  float  v8f;

union Frag { v16bf v; uint4 q[2]; };

static __device__ __forceinline__ unsigned short f32_to_bf16(float f) {
    union { float f; unsigned int u; } c; c.f = f;
    unsigned int u = c.u;
    u += 0x7FFFu + ((u >> 16) & 1u);          // round-to-nearest-even
    return (unsigned short)(u >> 16);
}

// A frag: 16 contiguous bf16 -> two b128 chunks at +0 and +8 ushorts
static __device__ __forceinline__ void load_a(Frag& f, const unsigned short* p) {
    f.q[0] = *(const uint4*)(p);
    f.q[1] = *(const uint4*)(p + 8);
}
// B frag: 16 contiguous bf16 -> two b128 chunks at +0 and +8 ushorts
static __device__ __forceinline__ void load_b(Frag& f, const unsigned short* p) {
    f.q[0] = *(const uint4*)(p);
    f.q[1] = *(const uint4*)(p + 8);
}

// ---------------------------------------------------------------------------
// Kernel 0: fp32 -> bf16 pre-conversion of Wh_w into workspace (run once;
// lets the GEMM stage B strips with global_load_async_to_lds_b128)
// ---------------------------------------------------------------------------
__global__ void cvt_kernel(const float* __restrict__ src,
                           unsigned short* __restrict__ dst, int n) {
    const int i = (blockIdx.x * 256 + threadIdx.x) * 4;
    if (i < n) {
        const float4 f = *(const float4*)(src + i);
        union { unsigned short u[4]; uint2 v; } o;
        o.u[0] = f32_to_bf16(f.x); o.u[1] = f32_to_bf16(f.y);
        o.u[2] = f32_to_bf16(f.z); o.u[3] = f32_to_bf16(f.w);
        *(uint2*)(dst + i) = o.v;
    }
}

// ---------------------------------------------------------------------------
// Kernel 1: per-batch aspect scalar
//   sv[b] = w_b + sum_c tanh( aspect[b,:] . Wv_w[c,:] + Wv_b[c] ) * w_w[H + c]
// ---------------------------------------------------------------------------
__global__ void sv_kernel(const float* __restrict__ aspect,
                          const float* __restrict__ Wv_w,
                          const float* __restrict__ Wv_b,
                          const float* __restrict__ w_w,
                          const float* __restrict__ w_b,
                          float* __restrict__ sv) {
    __shared__ float asp[AA];
    __shared__ float red[AA];
    const int b = blockIdx.x, t = threadIdx.x;     // blockDim.x == AA == 256
    asp[t] = aspect[b * AA + t];
    __syncthreads();
    float acc = Wv_b[t];
    const float* wr = Wv_w + (size_t)t * AA;
    #pragma unroll 8
    for (int a = 0; a < AA; ++a) acc += asp[a] * wr[a];
    red[t] = tanhf(acc) * w_w[HH + t];
    __syncthreads();
    for (int s = 128; s > 0; s >>= 1) {
        if (t < s) red[t] += red[t + s];
        __syncthreads();
    }
    if (t == 0) sv[b] = red[0] + w_b[0];
}

// ---------------------------------------------------------------------------
// Kernel 2: fused GEMM + tanh + weighted reduction (the WMMA kernel)
//   scores[b,s] = sv[b] + sum_g tanh( hidden[b,s,:] . Wh_w[g,:] + Wh_b[g] ) * w_w[g]
//
// 256 threads = 8 wave32 = 4 M-tiles x 2 N-groups; each wave owns two 16x16
// N-subtiles sharing one A fragment. BM=64 rows, B strip 64 cols.
// LDS: A 64x1024 bf16 (128KB, resident) + B 64x1024 bf16 (128KB) + 2x64 f32.
// K-loop is software-pipelined (double-buffered fragments) so ds_load_b128
// of iteration k+1 overlaps the two v_wmma of iteration k.
// B strips staged with global_load_async_to_lds_b128 when Whbf != nullptr.
// ---------------------------------------------------------------------------
__global__ void scores_kernel(const float* __restrict__ hidden,
                              const float* __restrict__ Wh_w,
                              const unsigned short* __restrict__ Whbf,
                              const float* __restrict__ Wh_b,
                              const float* __restrict__ w_w,
                              const float* __restrict__ sv,
                              float* __restrict__ scores) {
    extern __shared__ char smem[];
    unsigned short* Alds = (unsigned short*)smem;         // [64][1024] bf16
    unsigned short* Blds = Alds + 64 * HH;                // [64][1024] bf16
    float*          scr  = (float*)(Blds + 64 * HH);      // [2][64]

    const int tid   = threadIdx.x;
    const int lane  = tid & 31;
    const int wave  = tid >> 5;        // 0..7
    const int m_t   = wave & 3;        // M tile 0..3  (rows m_t*16 .. +15)
    const int n_t   = wave >> 2;       // N group 0..1 (strip cols n_t*32 .. +31)
    const int khalf = lane >> 4;       // 0/1 : lane half selects K group
    const int l15   = lane & 15;
    const int b     = blockIdx.y;
    const int s0    = blockIdx.x * 64;

    if (tid < 128) scr[tid] = 0.0f;

    // Stage A: hidden[b, s0:s0+64, :] -> bf16 LDS (row-major), once per block
    const float* hbase = hidden + ((size_t)b * SS + s0) * HH;
    for (int idx = tid; idx < 64 * HH; idx += 256)
        Alds[idx] = f32_to_bf16(hbase[idx]);
    __syncthreads();

    const unsigned short* arow  = Alds + (size_t)(m_t * 16 + l15) * HH + khalf * 16;
    const unsigned short* brow0 = Blds + (size_t)(n_t * 32 + l15) * HH + khalf * 16;
    const unsigned short* brow1 = brow0 + 16 * HH;
    const unsigned blds_off = (unsigned)(uintptr_t)(const void*)Blds; // LDS aperture: low 32 bits = LDS byte offset

    for (int n0 = 0; n0 < HH; n0 += 64) {
        // ---- Stage B strip: Wh_w[n0:n0+64, :] as bf16 into LDS ----
        if (Whbf) {
            // CDNA5 async copy engine: bf16 strip is already in global (ws),
            // issue 32 x b128 async-to-LDS per thread (GVS addressing).
            const unsigned short* strip = Whbf + (size_t)n0 * HH;
            #pragma unroll
            for (int i = 0; i < 32; ++i) {
                const unsigned o = (unsigned)(tid * 16 + i * 4096); // byte offset
                const unsigned lds_a = blds_off + o;
                asm volatile("global_load_async_to_lds_b128 %0, %1, %2"
                             :: "v"(lds_a), "v"(o), "s"(strip) : "memory");
            }
            asm volatile("s_wait_asynccnt 0" ::: "memory");
        } else {
            const float* wbase = Wh_w + (size_t)n0 * HH;
            for (int idx = tid; idx < 64 * HH; idx += 256)
                Blds[idx] = f32_to_bf16(wbase[idx]);
            if (n0 + 64 < HH)   // pull next strip toward L2 (global_prefetch_b8)
                __builtin_prefetch(Wh_w + (size_t)(n0 + 64) * HH + tid * 8, 0, 1);
        }
        __syncthreads();

        // ---- K loop: 2 WMMAs / step, double-buffered LDS fragments ----
        v8f acc0 = {};
        v8f acc1 = {};
        Frag a[2], b0[2], b1[2];
        load_a(a[0],  arow);
        load_b(b0[0], brow0);
        load_b(b1[0], brow1);
        for (int k0 = 0; k0 < HH; k0 += 32) {
            const int cur = (k0 >> 5) & 1;
            const int nxt = cur ^ 1;
            if (k0 + 32 < HH) {                      // prefetch next-k frags
                load_a(a[nxt],  arow  + k0 + 32);
                load_b(b0[nxt], brow0 + k0 + 32);
                load_b(b1[nxt], brow1 + k0 + 32);
            }
            acc0 = __builtin_amdgcn_wmma_f32_16x16x32_bf16(
                false, a[cur].v, false, b0[cur].v, (short)0, acc0, false, false);
            acc1 = __builtin_amdgcn_wmma_f32_16x16x32_bf16(
                false, a[cur].v, false, b1[cur].v, (short)0, acc1, false, false);
        }

        // ---- Fused epilogue: tanh + weight + 16-lane N reduction ----
        // acc*[v] = th[m, n], m = m_t*16 + v + 8*khalf,
        // n = n0 + n_t*32 + l15 (+16 for acc1)
        const int ng0 = n0 + n_t * 32 + l15;
        const float bias0 = Wh_b[ng0],      wn0 = w_w[ng0];
        const float bias1 = Wh_b[ng0 + 16], wn1 = w_w[ng0 + 16];
        #pragma unroll
        for (int v = 0; v < 8; ++v) {
            float val = tanhf(acc0[v] + bias0) * wn0
                      + tanhf(acc1[v] + bias1) * wn1;
            val += __shfl_xor(val, 1, 32);
            val += __shfl_xor(val, 2, 32);
            val += __shfl_xor(val, 4, 32);
            val += __shfl_xor(val, 8, 32);     // sum over 16-lane N group
            if (l15 == 0) {
                const int m = m_t * 16 + v + 8 * khalf;
                scr[n_t * 64 + m] += val;      // slot owned by this wave only
            }
        }
        __syncthreads();                        // B strip reuse + scr visibility
    }

    if (tid < 64)
        scores[(size_t)b * SS + s0 + tid] = scr[tid] + scr[64 + tid] + sv[b];
}

// ---------------------------------------------------------------------------
// Kernel 3: softmax over S, in place in the workspace scores buffer
// ---------------------------------------------------------------------------
__global__ void softmax_kernel(float* __restrict__ scores) {
    __shared__ float red[256];
    const int b = blockIdx.x, t = threadIdx.x;
    float* srow = scores + (size_t)b * SS;
    float m = -3.0e38f;
    for (int i = t; i < SS; i += 256) m = fmaxf(m, srow[i]);
    red[t] = m;
    __syncthreads();
    for (int s = 128; s > 0; s >>= 1) {
        if (t < s) red[t] = fmaxf(red[t], red[t + s]);
        __syncthreads();
    }
    m = red[0];
    __syncthreads();
    float sum = 0.0f;
    for (int i = t; i < SS; i += 256) {
        float e = __expf(srow[i] - m);
        srow[i] = e;
        sum += e;
    }
    red[t] = sum;
    __syncthreads();
    for (int s = 128; s > 0; s >>= 1) {
        if (t < s) red[t] += red[t + s];
        __syncthreads();
    }
    const float inv = 1.0f / red[0];
    for (int i = t; i < SS; i += 256) srow[i] *= inv;
}

// ---------------------------------------------------------------------------
// Kernel 4a: zero output (harness poisons d_out)
// ---------------------------------------------------------------------------
__global__ void zero_kernel(float* __restrict__ p, int n) {
    const int i = blockIdx.x * 256 + threadIdx.x;
    if (i < n) p[i] = 0.0f;
}

// ---------------------------------------------------------------------------
// Kernel 4b: r[b,h] = sum_s alpha[b,s] * hidden[b,s,h]   (memory-bound pass)
// grid (8 S-chunks, B); float4 per thread; f32 global atomics to combine.
// ---------------------------------------------------------------------------
__global__ void wsum_kernel(const float* __restrict__ hidden,
                            const float* __restrict__ alpha,
                            float* __restrict__ out) {
    const int b  = blockIdx.y;
    const int s0 = blockIdx.x * 256;
    const int t  = threadIdx.x;                 // owns h = 4t .. 4t+3
    float4 acc = make_float4(0.f, 0.f, 0.f, 0.f);
    const float* hb = hidden + ((size_t)b * SS + s0) * HH;
    const float* al = alpha + (size_t)b * SS + s0;
    for (int s = 0; s < 256; ++s) {
        const float a = al[s];
        const float4 h4 = *(const float4*)(hb + (size_t)s * HH + t * 4);
        acc.x += a * h4.x; acc.y += a * h4.y;
        acc.z += a * h4.z; acc.w += a * h4.w;
    }
    float* o = out + (size_t)b * HH + t * 4;
    unsafeAtomicAdd(o + 0, acc.x);
    unsafeAtomicAdd(o + 1, acc.y);
    unsafeAtomicAdd(o + 2, acc.z);
    unsafeAtomicAdd(o + 3, acc.w);
}

// ---------------------------------------------------------------------------
extern "C" void kernel_launch(void* const* d_in, const int* in_sizes, int n_in,
                              void* d_out, int out_size, void* d_ws, size_t ws_size,
                              hipStream_t stream) {
    const float* hidden = (const float*)d_in[0];   // [B,S,H]
    const float* aspect = (const float*)d_in[1];   // [B,1,A]
    const float* Wh_w   = (const float*)d_in[2];   // [H,H]
    const float* Wh_b   = (const float*)d_in[3];   // [H]
    const float* Wv_w   = (const float*)d_in[4];   // [A,A]
    const float* Wv_b   = (const float*)d_in[5];   // [A]
    const float* w_w    = (const float*)d_in[6];   // [1,H+A]
    const float* w_b    = (const float*)d_in[7];   // [1]
    float* out = (float*)d_out;                    // [B,1,H]

    // Workspace layout:
    //   [0,256)          sv[32] floats
    //   [256, 262400)    scores/alpha [B,S] floats
    //   [262400, +2MB)   Wh_w as bf16 (for async-to-LDS staging)
    const size_t SV_OFF  = 0;
    const size_t SC_OFF  = 256;
    const size_t WB_OFF  = 256 + (size_t)BB * SS * sizeof(float);   // 262400
    const size_t WS_NEED = WB_OFF + (size_t)HH * HH * sizeof(unsigned short);

    float* sv     = (float*)((char*)d_ws + SV_OFF);
    float* scores = (float*)((char*)d_ws + SC_OFF);
    unsigned short* Whbf =
        (ws_size >= WS_NEED) ? (unsigned short*)((char*)d_ws + WB_OFF) : nullptr;

    if (Whbf)
        cvt_kernel<<<dim3((HH * HH / 4 + 255) / 256), dim3(256), 0, stream>>>(
            Wh_w, Whbf, HH * HH);

    sv_kernel<<<dim3(BB), dim3(AA), 0, stream>>>(aspect, Wv_w, Wv_b, w_w, w_b, sv);

    const size_t smem = (size_t)(64 * HH + 64 * HH) * sizeof(unsigned short)
                      + 128 * sizeof(float);       // ~256.5 KB of 320 KB WGP LDS
    scores_kernel<<<dim3(SS / 64, BB), dim3(256), smem, stream>>>(
        hidden, Wh_w, Whbf, Wh_b, w_w, sv, scores);

    softmax_kernel<<<dim3(BB), dim3(256), 0, stream>>>(scores);

    zero_kernel<<<dim3((BB * HH + 255) / 256), dim3(256), 0, stream>>>(out, BB * HH);

    wsum_kernel<<<dim3(SS / 256, BB), dim3(256), 0, stream>>>(hidden, scores, out);
}
